// MultiHeadAttention_7490422964366
// MI455X (gfx1250) — matrix operand order repaired and checked by reading, not compiled, not run on hardware
//
#include <hip/hip_runtime.h>
#include <hip/hip_bf16.h>

typedef __attribute__((ext_vector_type(16))) __bf16 v16bf;
typedef __attribute__((ext_vector_type(8)))  float  v8f;
typedef __attribute__((ext_vector_type(4)))  __bf16 bf16x4;
typedef __attribute__((ext_vector_type(4)))  int    v4i;

#define S_LEN 2048
#define BATCH 2
#define DMODEL 1024
#define NHEAD 16
#define DK 64
#define SCALE 0.125f   // 1/sqrt(64)

struct Bytes32 { float4 lo, hi; };

// A-layout fragment: elements 0..7 at p[0..7], 8..15 at p[16..23]
__device__ __forceinline__ v16bf ld_frag_split(const __bf16* p) {
    Bytes32 b;
    b.lo = *(const float4*)(p);
    b.hi = *(const float4*)(p + 16);
    return __builtin_bit_cast(v16bf, b);
}
// B-layout fragment: 16 contiguous elements at p[0..15]
__device__ __forceinline__ v16bf ld_frag_contig(const __bf16* p) {
    Bytes32 b;
    b.lo = *(const float4*)(p);
    b.hi = *(const float4*)(p + 8);
    return __builtin_bit_cast(v16bf, b);
}

__device__ __forceinline__ v8f wmma_bf16(v16bf a, v16bf b, v8f c) {
    return __builtin_amdgcn_wmma_f32_16x16x32_bf16(false, a, false, b, (short)0, c, false, false);
}

// ---- CDNA5 async global->LDS DMA (ASYNCcnt path), guarded for compile-safety ----
#if defined(__gfx1250__) && __has_builtin(__builtin_amdgcn_global_load_async_to_lds_b128)
#define HAVE_ASYNC_LDS 1
typedef __attribute__((address_space(1))) v4i g_v4i;   // global (device) space
typedef __attribute__((address_space(3))) v4i l_v4i;   // LDS space
__device__ __forceinline__ void async_cp16(void* lds, const void* g) {
    __builtin_amdgcn_global_load_async_to_lds_b128((g_v4i*)g, (l_v4i*)lds, 0, 0);
}
__device__ __forceinline__ void wait_async0() {
#if __has_builtin(__builtin_amdgcn_s_wait_asynccnt)
    __builtin_amdgcn_s_wait_asynccnt(0);
#else
    asm volatile("s_wait_asynccnt 0x0" ::: "memory");
#endif
}
#else
#define HAVE_ASYNC_LDS 0
#endif

// ---------------------------------------------------------------------------
// Kernel 1: QKV projections.  C[t,e] = sum_d X[t,d]*W[e,d] + bias[e]
// grid = (N/128, M/128, 3), block = 128 (4 waves, each 64x64).
// Double-buffered LDS, global loads for step i+1 overlap WMMAs of step i.
// ---------------------------------------------------------------------------
#define LDS_ST 40   // padded row stride (bf16 elems): bank-conflict-free frags

__global__ __launch_bounds__(128) void proj_qkv_kernel(
    const float* __restrict__ q_in, const float* __restrict__ k_in, const float* __restrict__ v_in,
    const float* __restrict__ Wq, const float* __restrict__ Wk, const float* __restrict__ Wv,
    const float* __restrict__ bq, const float* __restrict__ bk, const float* __restrict__ bv,
    __bf16* __restrict__ Qh, __bf16* __restrict__ Kh, __bf16* __restrict__ Vt)
{
    __shared__ __bf16 ldsA[2][128 * LDS_ST];
    __shared__ __bf16 ldsB[2][128 * LDS_ST];

    const int tid  = threadIdx.x;
    const int lane = tid & 31;
    const int wid  = tid >> 5;
    const int z    = blockIdx.z;

    const float* X    = (z == 0) ? q_in : (z == 1) ? k_in : v_in;
    const float* W    = (z == 0) ? Wq   : (z == 1) ? Wk   : Wv;
    const float* bias = (z == 0) ? bq   : (z == 1) ? bk   : bv;

    const int n0 = blockIdx.x * 128;
    const int t0 = blockIdx.y * 128;
    const int waveM = wid & 1;
    const int waveN = wid >> 1;
    const int la = lane & 15;
    const int lh = lane >> 4;

    float4 aR[8], bR[8];
    auto ldTiles = [&](int k0) {
        const float* ar = X + (size_t)(t0 + tid) * DMODEL + k0;
        const float* br = W + (size_t)(n0 + tid) * DMODEL + k0;
#pragma unroll
        for (int c = 0; c < 8; c++) {
            aR[c] = *(const float4*)(ar + c * 4);
            bR[c] = *(const float4*)(br + c * 4);
        }
    };
    auto stTiles = [&](int buf) {
        __bf16* la_p = &ldsA[buf][tid * LDS_ST];
        __bf16* lb_p = &ldsB[buf][tid * LDS_ST];
#pragma unroll
        for (int c = 0; c < 8; c++) {
            bf16x4 ah, bh;
            ah.x = (__bf16)aR[c].x; ah.y = (__bf16)aR[c].y; ah.z = (__bf16)aR[c].z; ah.w = (__bf16)aR[c].w;
            bh.x = (__bf16)bR[c].x; bh.y = (__bf16)bR[c].y; bh.z = (__bf16)bR[c].z; bh.w = (__bf16)bR[c].w;
            *(bf16x4*)(la_p + c * 4) = ah;
            *(bf16x4*)(lb_p + c * 4) = bh;
        }
    };

    v8f acc[4][4] = {};

    ldTiles(0);
    stTiles(0);
    int cur = 0;
    for (int k0 = 0; k0 < DMODEL; k0 += 32) {
        __syncthreads();
        const bool more = (k0 + 32 < DMODEL);
        if (more) ldTiles(k0 + 32);          // overlap with WMMAs below

        v16bf afrag[4], bfrag[4];
#pragma unroll
        for (int mi = 0; mi < 4; mi++)
            afrag[mi] = ld_frag_split(&ldsA[cur][(waveM * 64 + mi * 16 + la) * LDS_ST + lh * 8]);
#pragma unroll
        for (int ni = 0; ni < 4; ni++)
            bfrag[ni] = ld_frag_split(&ldsB[cur][(waveN * 64 + ni * 16 + la) * LDS_ST + lh * 8]);
#pragma unroll
        for (int mi = 0; mi < 4; mi++)
#pragma unroll
            for (int ni = 0; ni < 4; ni++)
                acc[mi][ni] = wmma_bf16(afrag[mi], bfrag[ni], acc[mi][ni]);

        if (more) stTiles(cur ^ 1);
        cur ^= 1;
    }

    const int hi8 = lh * 8;
#pragma unroll
    for (int mi = 0; mi < 4; mi++) {
#pragma unroll
        for (int ni = 0; ni < 4; ni++) {
            const int e  = n0 + waveN * 64 + ni * 16 + la;
            const float be = bias[e];
            const int hh = e >> 6;      // head
            const int dk = e & 63;      // within-head dim
#pragma unroll
            for (int r = 0; r < 8; r++) {
                const int t = t0 + waveM * 64 + mi * 16 + r + hi8;
                const int s = t >> 1;   // B == 2
                const int b = t & 1;
                const float v = acc[mi][ni][r] + be;
                if (z == 2) {
                    Vt[((size_t)((b * NHEAD + hh) * DK + dk)) * S_LEN + s] = (__bf16)v;   // (B,H,DK,S)
                } else {
                    __bf16* dst = (z == 0) ? Qh : Kh;
                    dst[((size_t)((b * NHEAD + hh) * S_LEN + s)) * DK + dk] = (__bf16)v;  // (B,H,S,DK)
                }
            }
        }
    }
}

// ---------------------------------------------------------------------------
// Kernel 2: flash attention.  grid = (S/128, B*H), block = 128 (4 waves).
// Each wave owns 32 query rows; K/V double-buffered in LDS, next 64-key
// block staged via async global->LDS DMA while current block computes.
// ---------------------------------------------------------------------------
#define LDSK_ST 72
#define LDSP_ST 40

__global__ __launch_bounds__(128) void attn_kernel(
    const __bf16* __restrict__ Qh, const __bf16* __restrict__ Kh,
    const __bf16* __restrict__ Vt, __bf16* __restrict__ Xa)
{
    __shared__ __bf16 ldsK[2][64 * LDSK_ST];
    __shared__ __bf16 ldsV[2][64 * LDSK_ST];
    __shared__ __bf16 ldsP[4 * 32 * LDSP_ST];

    const int tid  = threadIdx.x;
    const int lane = tid & 31;
    const int wid  = tid >> 5;
    const int bh   = blockIdx.y;
    const int b    = bh >> 4;
    const int h    = bh & 15;
    const int i0   = blockIdx.x * 128;
    const int iw   = i0 + wid * 32;

    const __bf16* Qbh = Qh + (size_t)bh * S_LEN * DK;
    const __bf16* Kbh = Kh + (size_t)bh * S_LEN * DK;
    const __bf16* Vbh = Vt + (size_t)bh * DK * S_LEN;

    const int la  = lane & 15;
    const int lh  = lane >> 4;
    const int hi8 = lh * 8;

    auto stageKV = [&](int j0, int buf) {
        const int row = tid >> 1, half = tid & 1;
        const __bf16* ksrc = Kbh + (size_t)(j0 + row) * DK + half * 32;
        const __bf16* vsrc = Vbh + (size_t)row * S_LEN + j0 + half * 32;
        __bf16* kdst = &ldsK[buf][row * LDSK_ST + half * 32];
        __bf16* vdst = &ldsV[buf][row * LDSK_ST + half * 32];
#pragma unroll
        for (int c = 0; c < 32; c += 8) {
#if HAVE_ASYNC_LDS
            async_cp16(kdst + c, ksrc + c);
            async_cp16(vdst + c, vsrc + c);
#else
            *(float4*)(kdst + c) = *(const float4*)(ksrc + c);
            *(float4*)(vdst + c) = *(const float4*)(vsrc + c);
#endif
        }
    };

    // Q fragments straight from global (rows contiguous in dk)
    v16bf qf[2][2];
#pragma unroll
    for (int mi = 0; mi < 2; mi++)
#pragma unroll
        for (int ks = 0; ks < 2; ks++)
            qf[mi][ks] = ld_frag_split(Qbh + (size_t)(iw + mi * 16 + la) * DK + ks * 32 + lh * 8);

    v8f o[2][4] = {};
    float m_run[2][8], l_run[2][8];
#pragma unroll
    for (int mi = 0; mi < 2; mi++)
#pragma unroll
        for (int r = 0; r < 8; r++) { m_run[mi][r] = -__builtin_inff(); l_run[mi][r] = 0.0f; }

    const int jEnd = i0 + 128;   // causal: keys beyond block's last row are masked
    int buf = 0;
    stageKV(0, 0);
#if HAVE_ASYNC_LDS
    wait_async0();
#endif

    for (int j0 = 0; j0 < jEnd; j0 += 64) {
        __syncthreads();
        const bool more = (j0 + 64 < jEnd);
        if (more) stageKV(j0 + 64, buf ^ 1);   // DMA overlaps compute below

#pragma unroll
        for (int jl = 0; jl < 64; jl += 32) {
            // ---- scores: S = Q * K^T ----
            v8f sc[2][2] = {};
            v16bf kf[2][2];
#pragma unroll
            for (int nj = 0; nj < 2; nj++)
#pragma unroll
                for (int ks = 0; ks < 2; ks++)
                    kf[nj][ks] = ld_frag_split(&ldsK[buf][(jl + nj * 16 + la) * LDSK_ST + ks * 32 + lh * 8]);
#pragma unroll
            for (int mi = 0; mi < 2; mi++)
#pragma unroll
                for (int nj = 0; nj < 2; nj++)
#pragma unroll
                    for (int ks = 0; ks < 2; ks++)
                        sc[mi][nj] = wmma_bf16(qf[mi][ks], kf[nj][ks], sc[mi][nj]);

            // ---- causal mask + online softmax ----
#pragma unroll
            for (int mi = 0; mi < 2; mi++) {
#pragma unroll
                for (int r = 0; r < 8; r++) {
                    const int irow = iw + mi * 16 + r + hi8;
                    const int jc0  = j0 + jl + la;
                    float v0 = sc[mi][0][r] * SCALE;
                    float v1 = sc[mi][1][r] * SCALE;
                    if (jc0 > irow)      v0 = -1.0e9f;
                    if (jc0 + 16 > irow) v1 = -1.0e9f;
                    float mx = fmaxf(v0, v1);
                    mx = fmaxf(mx, __shfl_xor(mx, 1, 32));
                    mx = fmaxf(mx, __shfl_xor(mx, 2, 32));
                    mx = fmaxf(mx, __shfl_xor(mx, 4, 32));
                    mx = fmaxf(mx, __shfl_xor(mx, 8, 32));
                    const float mn = fmaxf(m_run[mi][r], mx);
                    const float a  = __expf(m_run[mi][r] - mn);
                    const float p0 = __expf(v0 - mn);
                    const float p1 = __expf(v1 - mn);
                    sc[mi][0][r] = p0;
                    sc[mi][1][r] = p1;
                    float ls = p0 + p1;
                    ls += __shfl_xor(ls, 1, 32);
                    ls += __shfl_xor(ls, 2, 32);
                    ls += __shfl_xor(ls, 4, 32);
                    ls += __shfl_xor(ls, 8, 32);
                    m_run[mi][r] = mn;
                    l_run[mi][r] = l_run[mi][r] * a + ls;
#pragma unroll
                    for (int nd = 0; nd < 4; nd++) o[mi][nd][r] *= a;
                }
            }

            // ---- re-swizzle P: C-layout -> A-layout via per-wave LDS bounce ----
            __bf16* Pw = &ldsP[wid * 32 * LDSP_ST];
#pragma unroll
            for (int mi = 0; mi < 2; mi++)
#pragma unroll
                for (int nj = 0; nj < 2; nj++)
#pragma unroll
                    for (int r = 0; r < 8; r++)
                        Pw[(mi * 16 + r + hi8) * LDSP_ST + nj * 16 + la] = (__bf16)sc[mi][nj][r];
            // same-wave DS ops are in-order: no barrier needed

            v16bf pf[2], vf[4];
#pragma unroll
            for (int mi = 0; mi < 2; mi++)
                pf[mi] = ld_frag_split(&Pw[(mi * 16 + la) * LDSP_ST + lh * 8]);
#pragma unroll
            for (int nd = 0; nd < 4; nd++)
                vf[nd] = ld_frag_split(&ldsV[buf][(nd * 16 + la) * LDSK_ST + jl + lh * 8]);
#pragma unroll
            for (int mi = 0; mi < 2; mi++)
#pragma unroll
                for (int nd = 0; nd < 4; nd++)
                    o[mi][nd] = wmma_bf16(pf[mi], vf[nd], o[mi][nd]);
        }

#if HAVE_ASYNC_LDS
        if (more) wait_async0();   // DMA for next block must land before next barrier
#endif
        buf ^= 1;
    }

    // ---- normalize + write attention output (token-major, d contiguous) ----
#pragma unroll
    for (int mi = 0; mi < 2; mi++) {
#pragma unroll
        for (int r = 0; r < 8; r++) {
            const int irow = iw + mi * 16 + r + hi8;
            const float inv = 1.0f / l_run[mi][r];
            const size_t base = ((size_t)(irow * BATCH + b)) * DMODEL + h * DK;
#pragma unroll
            for (int nd = 0; nd < 4; nd++)
                Xa[base + nd * 16 + la] = (__bf16)(o[mi][nd][r] * inv);
        }
    }
}

// ---------------------------------------------------------------------------
// Kernel 3: output projection.  out[t,e] = sum_d Xa[t,d]*Wo[e,d] + bo[e]  (f32)
// Same double-buffered pipeline; A is already bf16.
// ---------------------------------------------------------------------------
__global__ __launch_bounds__(128) void out_proj_kernel(
    const __bf16* __restrict__ Xa, const float* __restrict__ Wo,
    const float* __restrict__ bo, float* __restrict__ out)
{
    __shared__ __bf16 ldsA[2][128 * LDS_ST];
    __shared__ __bf16 ldsB[2][128 * LDS_ST];

    const int tid  = threadIdx.x;
    const int lane = tid & 31;
    const int wid  = tid >> 5;
    const int n0 = blockIdx.x * 128;
    const int t0 = blockIdx.y * 128;
    const int waveM = wid & 1;
    const int waveN = wid >> 1;
    const int la = lane & 15;
    const int lh = lane >> 4;

    uint4  aR[4];
    float4 bR[8];
    auto ldTiles = [&](int k0) {
        const __bf16* ar = Xa + (size_t)(t0 + tid) * DMODEL + k0;
        const float*  br = Wo + (size_t)(n0 + tid) * DMODEL + k0;
#pragma unroll
        for (int c = 0; c < 4; c++) aR[c] = *(const uint4*)(ar + c * 8);
#pragma unroll
        for (int c = 0; c < 8; c++) bR[c] = *(const float4*)(br + c * 4);
    };
    auto stTiles = [&](int buf) {
        __bf16* la_p = &ldsA[buf][tid * LDS_ST];
        __bf16* lb_p = &ldsB[buf][tid * LDS_ST];
#pragma unroll
        for (int c = 0; c < 4; c++) *(uint4*)(la_p + c * 8) = aR[c];
#pragma unroll
        for (int c = 0; c < 8; c++) {
            bf16x4 bh;
            bh.x = (__bf16)bR[c].x; bh.y = (__bf16)bR[c].y; bh.z = (__bf16)bR[c].z; bh.w = (__bf16)bR[c].w;
            *(bf16x4*)(lb_p + c * 4) = bh;
        }
    };

    v8f acc[4][4] = {};

    ldTiles(0);
    stTiles(0);
    int cur = 0;
    for (int k0 = 0; k0 < DMODEL; k0 += 32) {
        __syncthreads();
        const bool more = (k0 + 32 < DMODEL);
        if (more) ldTiles(k0 + 32);

        v16bf afrag[4], bfrag[4];
#pragma unroll
        for (int mi = 0; mi < 4; mi++)
            afrag[mi] = ld_frag_split(&ldsA[cur][(waveM * 64 + mi * 16 + la) * LDS_ST + lh * 8]);
#pragma unroll
        for (int ni = 0; ni < 4; ni++)
            bfrag[ni] = ld_frag_split(&ldsB[cur][(waveN * 64 + ni * 16 + la) * LDS_ST + lh * 8]);
#pragma unroll
        for (int mi = 0; mi < 4; mi++)
#pragma unroll
            for (int ni = 0; ni < 4; ni++)
                acc[mi][ni] = wmma_bf16(afrag[mi], bfrag[ni], acc[mi][ni]);

        if (more) stTiles(cur ^ 1);
        cur ^= 1;
    }

    const int hi8 = lh * 8;
#pragma unroll
    for (int mi = 0; mi < 4; mi++) {
#pragma unroll
        for (int ni = 0; ni < 4; ni++) {
            const int e = n0 + waveN * 64 + ni * 16 + la;
            const float be = bo[e];
#pragma unroll
            for (int r = 0; r < 8; r++) {
                const int t = t0 + waveM * 64 + mi * 16 + r + hi8;
                out[(size_t)t * DMODEL + e] = acc[mi][ni][r] + be;
            }
        }
    }
}

// ---------------------------------------------------------------------------
extern "C" void kernel_launch(void* const* d_in, const int* in_sizes, int n_in,
                              void* d_out, int out_size, void* d_ws, size_t ws_size,
                              hipStream_t stream) {
    const float* query = (const float*)d_in[0];
    const float* key   = (const float*)d_in[1];
    const float* value = (const float*)d_in[2];
    // d_in[3] = mask: causal (tril), applied analytically in-kernel
    const float* Wq = (const float*)d_in[4];
    const float* bq = (const float*)d_in[5];
    const float* Wk = (const float*)d_in[6];
    const float* bk = (const float*)d_in[7];
    const float* Wv = (const float*)d_in[8];
    const float* bv = (const float*)d_in[9];
    const float* Wo = (const float*)d_in[10];
    const float* bo = (const float*)d_in[11];

    char* ws = (char*)d_ws;
    const size_t SEG = (size_t)8 * 1024 * 1024;   // 8 MB per bf16 tensor
    __bf16* Qh = (__bf16*)(ws);
    __bf16* Kh = (__bf16*)(ws + SEG);
    __bf16* Vt = (__bf16*)(ws + 2 * SEG);
    __bf16* Xa = (__bf16*)(ws + 3 * SEG);

    proj_qkv_kernel<<<dim3(DMODEL / 128, (S_LEN * BATCH) / 128, 3), 128, 0, stream>>>(
        query, key, value, Wq, Wk, Wv, bq, bk, bv, Qh, Kh, Vt);

    attn_kernel<<<dim3(S_LEN / 128, BATCH * NHEAD), 128, 0, stream>>>(Qh, Kh, Vt, Xa);

    out_proj_kernel<<<dim3(DMODEL / 128, (S_LEN * BATCH) / 128), 128, 0, stream>>>(
        Xa, Wo, bo, (float*)d_out);
}
